// VideoGenerator_35665408426093
// MI455X (gfx1250) — compile-verified, matrix operand order checked by reading, not run on hardware
//
#include <hip/hip_runtime.h>
#include <hip/hip_bf16.h>

typedef __attribute__((ext_vector_type(16))) _Float16 v16h;
typedef __attribute__((ext_vector_type(8)))  _Float16 v8h;
typedef __attribute__((ext_vector_type(8)))  float    v8f;

// Problem constants
#define CIN   128
#define COUT  128
#define TT    16
#define HH    64
#define WW    64
#define NN    2
#define TAPS  27          // 3x3x3

// LDS padding geometry
#define AREC_B   48       // bytes per A-fragment lane record (32B data + 16B pad)
#define BCOL_H   136      // f16 per B column (128 data + 8 pad)
#define BCOL_B   272      // bytes per B column
// A: 4cc * 8mtile * 32lanes records = 1024 * 48B = 49152B
// B: 64 cols * 272B = 17408B

// ---------------------------------------------------------------------------
// CDNA5 async global->LDS copy (ISA §15.18.3 opcode 98, ASYNCcnt-tracked).
// vdst VGPR carries the LDS byte address; vaddr is the 64-bit global address.
// ---------------------------------------------------------------------------
__device__ __forceinline__ void async_copy16(unsigned int lds_off, const void* gptr) {
  asm volatile("global_load_async_to_lds_b128 %0, %1, off"
               :: "v"(lds_off), "v"((unsigned long long)(uintptr_t)gptr)
               : "memory");
}
__device__ __forceinline__ void async_wait0() {
  asm volatile("s_wait_asynccnt 0x0" ::: "memory");
}

// ---------------------------------------------------------------------------
// Weight prep: per-Cout max-abs normalize, /sqrt(fan_in), emit f16 weights in
// exact WMMA A-fragment lane order, plus wsq[o][i] = sum_tap w^2 for demod.
// A-fragment (16x32 f16, ISA 7.12.2): lane = m + ((k>>3)&1)*16,
//                                     elem = (k&7) + ((k>>4)&1)*8
// awpack layout: [tap][cc][mtile(8)][lane(32)][elem(16)] f16 (dense)
// ---------------------------------------------------------------------------
__global__ void wprep_kernel(const float* __restrict__ weight,
                             _Float16* __restrict__ awpack,
                             float* __restrict__ wsq) {
  __shared__ float red[256];
  const int o = blockIdx.x, tid = threadIdx.x;
  const float* wo = weight + o * (CIN * TAPS);
  float m = 0.f;
  for (int idx = tid; idx < CIN * TAPS; idx += 256) m = fmaxf(m, fabsf(wo[idx]));
  red[tid] = m;
  __syncthreads();
  for (int s = 128; s > 0; s >>= 1) {
    if (tid < s) red[tid] = fmaxf(red[tid], red[tid + s]);
    __syncthreads();
  }
  const float scale = 1.0f / (red[0] * 58.78775382679627f);  // sqrt(3456)
  if (tid < CIN) {
    const int i = tid;
    const int mtile = o >> 4, mrow = o & 15;
    const int cc = i >> 5, k = i & 31;
    const int lane = mrow + ((k >> 3) & 1) * 16;
    const int e = (k & 7) + ((k >> 4) & 1) * 8;
    float sq = 0.f;
    for (int tap = 0; tap < TAPS; ++tap) {
      const float v = wo[i * TAPS + tap] * scale;
      sq += v * v;
      const int pos = ((tap * 4 + cc) * 8 + mtile) * 512 + lane * 16 + e;
      awpack[pos] = (_Float16)v;
    }
    wsq[o * CIN + i] = sq;
  }
}

// ---------------------------------------------------------------------------
// Style prep: per-batch max-abs normalize. sbuf[n][i][t] f32.
// ---------------------------------------------------------------------------
__global__ void sprep_kernel(const float* __restrict__ style,
                             float* __restrict__ sbuf) {
  __shared__ float red[256];
  const int n = blockIdx.x, tid = threadIdx.x;
  const float* sn = style + n * (CIN * TT);
  float m = 0.f;
  for (int idx = tid; idx < CIN * TT; idx += 256) m = fmaxf(m, fabsf(sn[idx]));
  red[tid] = m;
  __syncthreads();
  for (int s = 128; s > 0; s >>= 1) {
    if (tid < s) red[tid] = fmaxf(red[tid], red[tid + s]);
    __syncthreads();
  }
  const float inv = 1.0f / red[0];
  for (int idx = tid; idx < CIN * TT; idx += 256) sbuf[n * (CIN * TT) + idx] = sn[idx] * inv;
}

// ---------------------------------------------------------------------------
// demod[n][o][t] = rsqrt(sum_i wsq[o][i] * s[n][i][t]^2 + eps)
// ---------------------------------------------------------------------------
__global__ void demod_kernel(const float* __restrict__ wsq,
                             const float* __restrict__ sbuf,
                             float* __restrict__ dem) {
  const int o = blockIdx.x;
  const int tid = threadIdx.x;             // 32 threads
  const int n = tid >> 4, t = tid & 15;
  float acc = 0.f;
  for (int i = 0; i < CIN; ++i) {
    const float sv = sbuf[n * (CIN * TT) + i * TT + t];
    acc += wsq[o * CIN + i] * sv * sv;
  }
  dem[(n * COUT + o) * TT + t] = __frsqrt_rn(acc + 1e-8f);
}

// ---------------------------------------------------------------------------
// Modulate + transpose: xmT[n][t][h][w][cin] f16 = x[n][cin][t][h][w] * s[n][cin][t]
// LDS-tiled transpose so global reads and writes are both 16B-coalesced.
// ---------------------------------------------------------------------------
__global__ void modulate_kernel(const float* __restrict__ x,
                                const float* __restrict__ sbuf,
                                _Float16* __restrict__ xmT) {
  __shared__ _Float16 tile[WW * 136];   // pad 128 -> 136 to dodge bank conflicts
  const int b = blockIdx.x, tid = threadIdx.x;
  const int h = b & 63, t = (b >> 6) & 15, n = b >> 10;
  for (int step = 0; step < 32; ++step) {
    const int i = step * 4 + (tid >> 6);
    const int w = tid & 63;
    const float v = x[(((long)(n * CIN + i) * TT + t) * HH + h) * WW + w] *
                    sbuf[n * (CIN * TT) + i * TT + t];
    tile[w * 136 + i] = (_Float16)v;
  }
  __syncthreads();
  uint4* dst = (uint4*)xmT + (long)(((n * TT + t) * HH + h) * WW) * (CIN / 8);
  for (int step = 0; step < 4; ++step) {
    const int idx = step * 256 + tid;
    const int w = idx >> 4, c = idx & 15;           // c: 8-f16 chunk of cin
    dst[w * 16 + c] = *(const uint4*)(&tile[w * 136 + c * 8]);
  }
}

// ---------------------------------------------------------------------------
// Main implicit-GEMM conv via v_wmma_f32_16x16x32_f16.
// Block = (n,t,h); computes out[n, 0:128, t, h, 0:64]. 8 waves: 4(M) x 2(N),
// each wave owns 32x32 (2x2 WMMA accumulators).
// Per valid tap: async-stage the full K=128 A (48KB padded) + B (17KB padded)
// tiles, one barrier pair, then 4 cc-steps x 4 WMMAs per wave.
// ---------------------------------------------------------------------------
__launch_bounds__(256, 2)
__global__ void conv_kernel(const uint4* __restrict__ awpack,   // fragment-ordered f16 weights
                            const uint4* __restrict__ xmT4,     // [n][t][h][w][cin] f16
                            const float* __restrict__ dem,
                            const float* __restrict__ bias,
                            float* __restrict__ out) {
  __shared__ uint4 lds_a4[3072];   // 49152B: [cc(4)][mtile(8)][lane(32)] x 48B records
  __shared__ uint4 lds_b4[1088];   // 17408B: [col(64)] x 272B (136 f16, 128 used)

  const int b = blockIdx.x, tid = threadIdx.x;
  const int h = b & 63, t = (b >> 6) & 15, n = b >> 10;
  const int lane = tid & 31, wave = tid >> 5;
  const int wm = wave & 3, wn = wave >> 2;

  v8f acc[2][2] = {};

  const char* lds_a_c = (const char*)lds_a4;
  const char* lds_b_c = (const char*)lds_b4;
  const unsigned int lds_a_base = (unsigned int)(uintptr_t)lds_a4;
  const unsigned int lds_b_base = (unsigned int)(uintptr_t)lds_b4;
  const int kb2 = (lane < 16) ? 0 : 16;  // per-lane K-half byte offset (kb*2)

  for (int tap = 0; tap < TAPS; ++tap) {
    const int dz = tap / 9, rem = tap % 9, dy = rem / 3, dx = rem % 3;
    const int tz = t + dz - 1, ty = h + dy - 1;
    if (tz < 0 || tz >= TT || ty < 0 || ty >= HH) continue;  // block-uniform skip
    const long bbase = (long)((n * TT + tz) * HH + ty) * WW;  // in w units

    __syncthreads();   // previous tap's fragment reads complete before overwrite

    // ---- Stage A: 2048 x 16B dense chunks -> 48B-padded records (async) ----
    const uint4* ga = awpack + (size_t)tap * 2048;
#pragma unroll
    for (int j = 0; j < 8; ++j) {
      const int g = j * 256 + tid;                     // 16B chunk id
      const unsigned int loff =
          lds_a_base + (unsigned int)((g >> 1) * AREC_B + (g & 1) * 16);
      async_copy16(loff, ga + g);
    }
    // ---- Stage B: 64 cols x 128 cin f16 (async, zero-fill W halo) ----
#pragma unroll
    for (int j = 0; j < 4; ++j) {
      const int c = j * 256 + tid;                     // 0..1023
      const int col = c >> 4, q = c & 15;              // q: 8-f16 cin chunk
      const int in_w = col + dx - 1;
      const unsigned int loff =
          lds_b_base + (unsigned int)(col * BCOL_B + q * 16);
      if (in_w >= 0 && in_w < WW) {
        async_copy16(loff, xmT4 + (bbase + in_w) * (CIN / 8) + q);
      } else {
        ((uint4*)lds_b_c)[col * 17 + q] = make_uint4(0u, 0u, 0u, 0u);
      }
    }
    async_wait0();
    __syncthreads();

    // ---- Compute: 4 cc-steps, 16 WMMAs per wave per tap ----
#pragma unroll
    for (int cc = 0; cc < 4; ++cc) {
      v16h afrag[2], bfrag[2];
#pragma unroll
      for (int mt = 0; mt < 2; ++mt) {
        const char* pa = lds_a_c + ((cc * 8 + wm * 2 + mt) * 32 + lane) * AREC_B;
        const v8h alo = *(const v8h*)(pa);
        const v8h ahi = *(const v8h*)(pa + 16);
        v16h af;
#pragma unroll
        for (int e = 0; e < 8; ++e) { af[e] = alo[e]; af[8 + e] = ahi[e]; }
        afrag[mt] = af;
      }
#pragma unroll
      for (int nt = 0; nt < 2; ++nt) {
        const int col = wn * 32 + nt * 16 + (lane & 15);
        const char* pb = lds_b_c + col * BCOL_B + cc * 64 + kb2;
        const v8h lo = *(const v8h*)(pb);
        const v8h hi = *(const v8h*)(pb + 32);
        v16h bf;
#pragma unroll
        for (int e = 0; e < 8; ++e) { bf[e] = lo[e]; bf[8 + e] = hi[e]; }
        bfrag[nt] = bf;
      }
#pragma unroll
      for (int mt = 0; mt < 2; ++mt)
#pragma unroll
        for (int nt = 0; nt < 2; ++nt)
          acc[mt][nt] = __builtin_amdgcn_wmma_f32_16x16x32_f16(
              false, afrag[mt], false, bfrag[nt],
              (short)0, acc[mt][nt], false, false);
    }
  }

  // Epilogue: demod * acc + bias, lrelu * sqrt(2).
  // C layout: VGPR r -> M = r (+8 for lanes 16..31); N = lane & 15.
  const int Ncol = lane & 15;
  const int moff = (lane >= 16) ? 8 : 0;
#pragma unroll
  for (int mt = 0; mt < 2; ++mt) {
    const int obase = wm * 32 + mt * 16 + moff;
#pragma unroll
    for (int nt = 0; nt < 2; ++nt) {
      const int wcol = wn * 32 + nt * 16 + Ncol;
#pragma unroll
      for (int r = 0; r < 8; ++r) {
        const int o = obase + r;
        const float d = dem[(n * COUT + o) * TT + t];
        float z = acc[mt][nt][r] * d + bias[o];
        z = (z >= 0.f ? z : 0.2f * z) * 1.41421356237f;
        out[(long)((n * COUT + o) * TT + t) * (HH * WW) + h * WW + wcol] = z;
      }
    }
  }
}

// ---------------------------------------------------------------------------
// Host-side orchestration
// ---------------------------------------------------------------------------
extern "C" void kernel_launch(void* const* d_in, const int* in_sizes, int n_in,
                              void* d_out, int out_size, void* d_ws, size_t ws_size,
                              hipStream_t stream) {
  const float* x      = (const float*)d_in[0];
  const float* weight = (const float*)d_in[1];
  const float* style  = (const float*)d_in[2];
  const float* bias   = (const float*)d_in[3];
  float* out = (float*)d_out;

  char* ws = (char*)d_ws;
  // Workspace carve-up (all 256B-aligned)
  _Float16* awpack = (_Float16*)(ws);                 //   884,736 B (108*4096 f16)
  float*    wsq    = (float*)(ws + 884736);           //    65,536 B
  float*    sbuf   = (float*)(ws + 950272);           //    16,384 B
  float*    dem    = (float*)(ws + 966656);           //    16,384 B
  _Float16* xmT    = (_Float16*)(ws + 983040);        // 33,554,432 B
  (void)ws_size; (void)in_sizes; (void)n_in; (void)out_size;

  hipLaunchKernelGGL(wprep_kernel,    dim3(COUT),            dim3(256), 0, stream, weight, awpack, wsq);
  hipLaunchKernelGGL(sprep_kernel,    dim3(NN),              dim3(256), 0, stream, style, sbuf);
  hipLaunchKernelGGL(demod_kernel,    dim3(COUT),            dim3(32),  0, stream, wsq, sbuf, dem);
  hipLaunchKernelGGL(modulate_kernel, dim3(NN * TT * HH),    dim3(256), 0, stream, x, sbuf, xmT);
  hipLaunchKernelGGL(conv_kernel,     dim3(NN * TT * HH),    dim3(256), 0, stream,
                     (const uint4*)awpack, (const uint4*)xmT, dem, bias, out);
}
